// LBConv_11682311045440
// MI455X (gfx1250) — matrix operand order repaired and checked by reading, not compiled, not run on hardware
//
#include <hip/hip_runtime.h>
#include <stdint.h>

// ---------------------------------------------------------------------------
// MI455X (gfx1250) fused LBConv:
//   stage1: 3x3 ternary conv (256->512) + ReLU  == GEMM [512x2304]x[2304x100352]
//   stage2: 1x1 conv (512->256) + bias          == GEMM [256x512]x[512x100352]
// Compute-bound (~263 GFLOP vs ~210MB HBM) -> bf16 WMMA, f32 acc.
// K reordered as g = (kh*3+kw)*256 + cin so each 32-wide K chunk has uniform
// (kh,kw) and contiguous cin; x pre-transposed to NHWC bf16 so the whole
// im2col B stage is ONE async b128 global->LDS copy per thread per K-step.
// A/B tiles double-buffered through the ASYNCcnt path to overlap with WMMA.
// ---------------------------------------------------------------------------

typedef __attribute__((ext_vector_type(16))) __bf16 bf16x16;
typedef __attribute__((ext_vector_type(8)))  float  v8f;

#define B_     32
#define CIN_   256
#define H_     56
#define W_     56
#define COUT_  256
#define CMID_  512
#define KK_    2304          // CIN * 9
#define HW_    3136          // H*W
#define NTILE  64
#define TILES_PER_IMG (HW_ / NTILE)   // 49

// Native RTNE f32->bf16 (single v_cvt on gfx1250)
__device__ __forceinline__ unsigned short f2bf(float f) {
  union { __bf16 h; unsigned short u; } c;
  c.h = (__bf16)f;
  return c.u;
}

// x: fp32 NCHW -> bf16 NHWC.  Block = (b*H + h), thread = channel.
// Writes are 512B-coalesced across lanes; reads stream rows through cache.
__global__ void cvt_x_nhwc(const float* __restrict__ in,
                           unsigned short* __restrict__ out) {
  int bh = blockIdx.x;                 // 0 .. B*H-1
  int b  = bh / H_;
  int h  = bh - b * H_;
  int c  = threadIdx.x;                // 0..255
  const float* src = &in[(((size_t)b * CIN_ + c) * H_ + h) * W_];
  unsigned short* dst = &out[(size_t)bh * W_ * CIN_ + c];
  for (int w = 0; w < W_; ++w)
    dst[(size_t)w * CIN_] = f2bf(src[w]);
}

// binary_w: [CMID][cin*9 + pos] fp32 -> bf16 [CMID][pos*256 + cin]
__global__ void cvt_w1_reord(const float* __restrict__ in,
                             unsigned short* __restrict__ out) {
  int o = blockIdx.x * blockDim.x + threadIdx.x;
  if (o >= CMID_ * KK_) return;
  int cmid = o / KK_;
  int r    = o - cmid * KK_;
  int pos  = r >> 8;
  int cin  = r & 255;
  out[o] = f2bf(in[(size_t)cmid * KK_ + cin * 9 + pos]);
}

__global__ void cvt_f32_bf16(const float* __restrict__ in,
                             unsigned short* __restrict__ out, int n) {
  int i = blockIdx.x * blockDim.x + threadIdx.x;
  if (i < n) out[i] = f2bf(in[i]);
}

// Async global->LDS 16B copy (CDNA5 VGLOBAL async op, tracked by ASYNCcnt).
__device__ __forceinline__ void async_g2l_b128(unsigned lds_off,
                                               const void* gaddr) {
  asm volatile("global_load_async_to_lds_b128 %0, %1, off"
               :: "v"(lds_off), "v"(gaddr) : "memory");
}
__device__ __forceinline__ void wait_async0() {
  asm volatile("s_wait_asynccnt 0x0" ::: "memory");
}

// Fragment = 16 bf16 = 8 VGPRs; filled with two aligned 16B LDS loads.
union Frag { bf16x16 v; uint4 q[2]; };

__global__ __launch_bounds__(256)
void lbconv_fused(const unsigned short* __restrict__ xbf,   // NHWC bf16
                  const unsigned short* __restrict__ w1bf,  // [CMID][KK_] reord
                  const unsigned short* __restrict__ fcbf,  // [COUT][CMID]
                  const float*          __restrict__ fcb,   // [COUT]
                  float*                __restrict__ out)   // [B][COUT][H][W]
{
  extern __shared__ __align__(16) unsigned short smem[];
  unsigned short* Yt  = smem;                    // [64][512] bf16 (64 KB)
  unsigned short* As0 = smem + NTILE * CMID_;    // [512][32] bf16 (32 KB)
  unsigned short* As1 = As0 + CMID_ * 32;        // [512][32] bf16 (32 KB)
  unsigned short* Bs0 = As1 + CMID_ * 32;        // [64][32]  bf16 ( 4 KB)
  unsigned short* Bs1 = Bs0 + NTILE * 32;        // [64][32]  bf16 ( 4 KB)

  const int wg  = blockIdx.x;
  const int b   = wg / TILES_PER_IMG;
  const int hw0 = (wg % TILES_PER_IMG) * NTILE;

  const int tid  = threadIdx.x;
  const int lane = tid & 31;
  const int wave = tid >> 5;
  const int half = lane >> 4;   // K-half selector for 16-bit A/B layouts
  const int l16  = lane & 15;   // M (A) or N (B/C) index within tile

  // =========================== Phase 1: 3x3 conv ==========================
  const int m0 = (wave >> 1) * 32;
  const int n0 = (wave & 1) * 32;

  // per-thread im2col geometry (K chunk has uniform kh,kw)
  const int p   = tid >> 2;            // pixel 0..63
  const int c8  = (tid & 3) * 8;       // cin sub-offset
  const int hw  = hw0 + p;
  const int h   = hw / W_;
  const int w   = hw - h * W_;
  const long long base0 =
      (((long long)b * H_ + (h - 1)) * W_ + (w - 1)) * CIN_;  // may be "neg"
  const unsigned bsrel = (unsigned)(p * 32 + c8) * 2;          // bytes in tile

  // stage phase-1 tiles for K-chunk kc into (Asb, Bsb)
  auto stage1 = [&](int kc, unsigned short* Asb, unsigned short* Bsb) {
    // A: weights [512 x 32], 2 rows / thread, 4 x async b128 each
    int row0 = tid * 2;
#pragma unroll
    for (int rr = 0; rr < 2; ++rr) {
      int row = row0 + rr;
      const unsigned short* g = &w1bf[(size_t)row * KK_ + kc];
      unsigned lo = (unsigned)(size_t)(void*)&Asb[row * 32];
#pragma unroll
      for (int c = 0; c < 4; ++c)
        async_g2l_b128(lo + c * 16, g + c * 8);
    }
    // B: one async b128 per thread (8 contiguous cin in NHWC), OOB -> zeros
    int pos = kc >> 8;                               // uniform (SALU)
    int kh  = pos >= 6 ? 2 : (pos >= 3 ? 1 : 0);
    int kw  = pos - 3 * kh;
    int cin0 = kc & 255;
    int ih = h - 1 + kh;
    int iw = w - 1 + kw;
    unsigned blo = (unsigned)(size_t)(void*)Bsb + bsrel;
    if ((unsigned)ih < (unsigned)H_ && (unsigned)iw < (unsigned)W_) {
      const unsigned short* ga =
          xbf + base0 + (kh * W_ + kw) * CIN_ + cin0 + c8;
      async_g2l_b128(blo, ga);
    } else {
      uint4 z = {0u, 0u, 0u, 0u};
      *reinterpret_cast<uint4*>(&Bsb[p * 32 + c8]) = z;
    }
  };

  v8f acc[4][2][2] = {};   // [midgroup][tm][tn] = 128 VGPRs/lane

  stage1(0, As0, Bs0);
  for (int kc = 0; kc < KK_; kc += 32) {
    wait_async0();
    __syncthreads();
    unsigned short* Asc = ((kc >> 5) & 1) ? As1 : As0;
    unsigned short* Bsc = ((kc >> 5) & 1) ? Bs1 : Bs0;
    if (kc + 32 < KK_) {
      unsigned short* Asn = ((kc >> 5) & 1) ? As0 : As1;
      unsigned short* Bsn = ((kc >> 5) & 1) ? Bs0 : Bs1;
      stage1(kc + 32, Asn, Bsn);     // async DMA overlaps WMMA below
    }

    Frag fbb[2];
#pragma unroll
    for (int tn = 0; tn < 2; ++tn) {
      int n = n0 + tn * 16 + l16;
      fbb[tn].q[0] = *reinterpret_cast<const uint4*>(&Bsc[n * 32 + half * 8]);
      fbb[tn].q[1] = *reinterpret_cast<const uint4*>(&Bsc[n * 32 + 16 + half * 8]);
    }
#pragma unroll
    for (int mc = 0; mc < 4; ++mc) {
      Frag fa[2];
#pragma unroll
      for (int tm = 0; tm < 2; ++tm) {
        int m = mc * 128 + m0 + tm * 16 + l16;
        fa[tm].q[0] = *reinterpret_cast<const uint4*>(&Asc[m * 32 + half * 8]);
        fa[tm].q[1] = *reinterpret_cast<const uint4*>(&Asc[m * 32 + 16 + half * 8]);
      }
#pragma unroll
      for (int tm = 0; tm < 2; ++tm)
#pragma unroll
        for (int tn = 0; tn < 2; ++tn)
          acc[mc][tm][tn] = __builtin_amdgcn_wmma_f32_16x16x32_bf16(
              false, fa[tm].v, false, fbb[tn].v, (short)0, acc[mc][tm][tn],
              false, false);
    }
  }

  // ---- ReLU + pack to bf16 Y-tile [pixel][mid] (K-contiguous for phase 2)
#pragma unroll
  for (int mc = 0; mc < 4; ++mc)
#pragma unroll
    for (int tm = 0; tm < 2; ++tm)
#pragma unroll
      for (int tn = 0; tn < 2; ++tn) {
        int n  = n0 + tn * 16 + l16;
        int ms = mc * 128 + m0 + tm * 16 + half * 8;
        union { unsigned short e[8]; uint4 q; } pk;
#pragma unroll
        for (int r = 0; r < 8; ++r) {
          float v = acc[mc][tm][tn][r];
          pk.e[r] = f2bf(v > 0.f ? v : 0.f);
        }
        *reinterpret_cast<uint4*>(&Yt[n * CMID_ + ms]) = pk.q;
      }

  // ====================== Phase 2: 1x1 conv + bias ========================
  const int m0b = wave * 32;
  v8f acc2[2][4] = {};

  auto stage2 = [&](int kc, unsigned short* Asb) {
    const unsigned short* g = &fcbf[(size_t)tid * CMID_ + kc];
    unsigned lo = (unsigned)(size_t)(void*)&Asb[tid * 32];
#pragma unroll
    for (int c = 0; c < 4; ++c)
      async_g2l_b128(lo + c * 16, g + c * 8);
  };

  stage2(0, As0);
  for (int kc = 0; kc < CMID_; kc += 32) {
    wait_async0();
    __syncthreads();
    unsigned short* Asc = ((kc >> 5) & 1) ? As1 : As0;
    if (kc + 32 < CMID_)
      stage2(kc + 32, ((kc >> 5) & 1) ? As0 : As1);

    Frag fa[2], fb[4];
#pragma unroll
    for (int tm = 0; tm < 2; ++tm) {
      int m = m0b + tm * 16 + l16;
      fa[tm].q[0] = *reinterpret_cast<const uint4*>(&Asc[m * 32 + half * 8]);
      fa[tm].q[1] = *reinterpret_cast<const uint4*>(&Asc[m * 32 + 16 + half * 8]);
    }
#pragma unroll
    for (int tn = 0; tn < 4; ++tn) {
      int n = tn * 16 + l16;
      fb[tn].q[0] =
          *reinterpret_cast<const uint4*>(&Yt[n * CMID_ + kc + half * 8]);
      fb[tn].q[1] =
          *reinterpret_cast<const uint4*>(&Yt[n * CMID_ + kc + 16 + half * 8]);
    }
#pragma unroll
    for (int tm = 0; tm < 2; ++tm)
#pragma unroll
      for (int tn = 0; tn < 4; ++tn)
        acc2[tm][tn] = __builtin_amdgcn_wmma_f32_16x16x32_bf16(
            false, fa[tm].v, false, fb[tn].v, (short)0, acc2[tm][tn],
            false, false);
  }

  // ---- epilogue: bias add + fp32 store (NCHW) ----
#pragma unroll
  for (int tm = 0; tm < 2; ++tm) {
    int cob = m0b + tm * 16 + half * 8;
#pragma unroll
    for (int tn = 0; tn < 4; ++tn) {
      int n = tn * 16 + l16;
      size_t base = (size_t)b * COUT_ * HW_ + (size_t)(hw0 + n);
#pragma unroll
      for (int r = 0; r < 8; ++r) {
        int co = cob + r;
        out[base + (size_t)co * HW_] = acc2[tm][tn][r] + fcb[co];
      }
    }
  }
}

extern "C" void kernel_launch(void* const* d_in, const int* in_sizes, int n_in,
                              void* d_out, int out_size, void* d_ws,
                              size_t ws_size, hipStream_t stream) {
  const float* x   = (const float*)d_in[0];  // [32,256,56,56]
  const float* w1  = (const float*)d_in[1];  // [512,256,3,3]
  const float* fcw = (const float*)d_in[2];  // [256,512,1,1]
  const float* fcb = (const float*)d_in[3];  // [256]
  float* out = (float*)d_out;

  const int NX  = B_ * CIN_ * H_ * W_;   // 25,690,112
  const int NW1 = CMID_ * KK_;           //  1,179,648
  const int NFC = COUT_ * CMID_;         //    131,072

  unsigned short* xbf  = (unsigned short*)d_ws;   // NHWC bf16
  unsigned short* w1bf = xbf + NX;                // K-reordered bf16
  unsigned short* fcbf = w1bf + NW1;

  cvt_x_nhwc<<<B_ * H_, CIN_, 0, stream>>>(x, xbf);
  cvt_w1_reord<<<(NW1 + 255) / 256, 256, 0, stream>>>(w1, w1bf);
  cvt_f32_bf16<<<(NFC + 255) / 256, 256, 0, stream>>>(fcw, fcbf, NFC);

  const size_t smem_bytes =
      (size_t)(NTILE * CMID_ + 2 * CMID_ * 32 + 2 * NTILE * 32) *
      sizeof(unsigned short);
  // 64KB Y-tile + 2x32KB A + 2x4KB B = 136KB (WGP has 320KB LDS)
  lbconv_fused<<<B_ * TILES_PER_IMG, 256, smem_bytes, stream>>>(
      xbf, w1bf, fcbf, fcb, out);
}
